// Net_15762529976340
// MI455X (gfx1250) — matrix operand order repaired and checked by reading, not compiled, not run on hardware
//
#include <hip/hip_runtime.h>

typedef __attribute__((ext_vector_type(16))) _Float16 v16h;
typedef __attribute__((ext_vector_type(8)))  _Float16 v8h;
typedef __attribute__((ext_vector_type(8)))  float    v8f;

#define NCLS 21

// ---------------------------------------------------------------- conversions
__global__ __launch_bounds__(256) void cvt_f32_to_f16(const float* __restrict__ in,
                                                      _Float16* __restrict__ out, int n) {
    int i = blockIdx.x * 256 + threadIdx.x;
    if (i < n) out[i] = (_Float16)in[i];
}

// weights f32 OIHW [Cout, Cin, 3, 3] -> f16 [Cout, 9, Cinp] with K reordered to
// k = r*Cinp + ci and ci zero-padded to Cinp (multiple of 32). This makes every
// 32-wide GEMM K-step single-r and branch-free in the conv hot loop.
__global__ __launch_bounds__(256) void cvt_wgt_reorder(const float* __restrict__ in,
                                                       _Float16* __restrict__ out,
                                                       int Cin, int Cinp, int Cout) {
    int Kpad = 9 * Cinp;
    int total = Cout * Kpad;
    int i = blockIdx.x * 256 + threadIdx.x;
    if (i >= total) return;
    int co = i / Kpad;
    int k  = i - co * Kpad;
    int r  = k / Cinp;
    int ci = k - r * Cinp;
    _Float16 v = (_Float16)0.0f;
    if (ci < Cin) v = (_Float16)in[((size_t)co * Cin + ci) * 9 + r];
    out[i] = v;
}

// ---------------------------------------------------------------- conv3x3 SAME + bias + relu
// Implicit GEMM: D[Cout, B*H*W] = W[Cout, 9*Cinp] x Im2col[9*Cinp, B*H*W]
// MT x 64 tile / 256-thread block, K-steps of 32 via v_wmma_f32_16x16x32_f16,
// double-buffered LDS, one barrier per K-step. K order = (r, ci): halo math is
// hoisted to uniform r-boundaries; hot-path gather is add+load+ds_store.
template<int MT>
__global__ __launch_bounds__(256) void conv3x3_wmma(const _Float16* __restrict__ in,
                                                    const _Float16* __restrict__ wgt,
                                                    const float* __restrict__ bias,
                                                    _Float16* __restrict__ out,
                                                    int Cin, int Cout, int H, int W)
{
    constexpr int NACC = (MT == 128) ? 4 : 2;
    const int Cinp = (Cin + 31) & ~31;
    const int Kpad = 9 * Cinp;
    const int HW   = H * W;
    const int n0 = blockIdx.x * 64;   // spatial tile
    const int m0 = blockIdx.y * MT;   // Cout tile

    __shared__ __align__(32) _Float16 lds_a [2][MT * 32];  // [m][k]
    __shared__ __align__(32) _Float16 lds_bt[2][64 * 32];  // [n][k] transposed im2col

    const int t    = threadIdx.x;
    const int lane = t & 31;
    const int wave = t >> 5;
    const int l16 = lane & 15;
    const int hi  = lane >> 4;
    const int kbA = hi ? 16 : 0;                      // A-frag K base (contiguous k = 16*hi + s)
    const int kbB = hi ? 16 : 0;                      // B-frag K base (contiguous k = 16*hi + s)
    const int waveM = (MT == 128) ? wave : (wave & 3);
    const int nsb   = (MT == 128) ? 0    : 2 * (wave >> 2);
    const int aoffA = (waveM * 16 + l16) * 32;

    // ---- A-tile source/dest: unconditional 16B load(s) per thread per step
    const _Float16* asrc0 = wgt + (size_t)(m0 + (t >> 2)) * Kpad + (t & 3) * 8;
    const _Float16* asrc1 = asrc0 + (size_t)64 * Kpad;  // only used when MT==128
    const int adst0 = (t >> 2) * 32 + (t & 3) * 8;
    const int adst1 = adst0 + 64 * 32;

    // ---- B-tile: all 8 slots of this thread share nn = t&63; kk_i = 4i + (t>>6)
    const int nn = t & 63;
    const int t6 = t >> 6;
    const int ng  = n0 + nn;
    const int b   = ng / HW;
    const int rem = ng - b * HW;
    const int y   = rem / W;
    const int x   = rem - y * W;
    const int basek0 = b * Cin * HW + y * W + x + t6 * HW;
    const int ldsb   = nn * 32 + t6;
    const int HW4    = 4 * HW;

    int lr = 0, lcc = 0, offr = 0;
    bool ok = false;

    auto load_tiles = [&](int buf) {
        if (lcc == 0) {                      // uniform: new filter tap r
            int q  = lr / 3;
            int dy = q - 1;
            int dx = lr - q * 3 - 1;
            offr = dy * W + dx;
            int yy = y + dy, xx = x + dx;
            ok = ((unsigned)yy < (unsigned)H) && ((unsigned)xx < (unsigned)W);
        }
        *(v8h*)&lds_a[buf][adst0] = *(const v8h*)asrc0;
        __builtin_prefetch(asrc0 + 32, 0, 3);          // global_prefetch next chunk
        asrc0 += 32;
        if constexpr (MT == 128) {
            *(v8h*)&lds_a[buf][adst1] = *(const v8h*)asrc1;
            asrc1 += 32;
        }
        const int abase = basek0 + offr + lcc * HW;
        #pragma unroll
        for (int i = 0; i < 8; ++i) {
            int ci = lcc + 4 * i + t6;
            _Float16 v = (_Float16)0.0f;
            if (ok && ci < Cin)
                v = in[abase + i * HW4];
            lds_bt[buf][ldsb + 4 * i] = v;
        }
        lcc += 32;
        if (lcc >= Cinp) { lcc = 0; ++lr; }
    };

    v8f acc[NACC] = {};
    const int nsteps = Kpad >> 5;

    load_tiles(0);                                     // prologue: tile 0
    for (int s = 0; s < nsteps; ++s) {
        __syncthreads();                               // tile s visible; other buf free
        if (s + 1 < nsteps) load_tiles((s + 1) & 1);

        const int buf = s & 1;
        v16h a;
        {
            v8h alo = *(const v8h*)&lds_a[buf][aoffA + kbA];
            v8h ahi = *(const v8h*)&lds_a[buf][aoffA + 8 + kbA];
            #pragma unroll
            for (int j = 0; j < 8; ++j) { a[j] = alo[j]; a[8 + j] = ahi[j]; }
        }
        #pragma unroll
        for (int j = 0; j < NACC; ++j) {
            v16h bf = *(const v16h*)&lds_bt[buf][((nsb + j) * 16 + l16) * 32 + kbB];
            acc[j] = __builtin_amdgcn_wmma_f32_16x16x32_f16(false, a, false, bf,
                                                            (short)0, acc[j], false, false);
        }
    }

    // ---- epilogue: bias + relu, store f16 NCHW
    #pragma unroll
    for (int j = 0; j < NACC; ++j) {
        int ngj  = n0 + (nsb + j) * 16 + l16;
        int bb   = ngj / HW;
        int remj = ngj - bb * HW;
        int yj   = remj / W;
        int xj   = remj - yj * W;
        #pragma unroll
        for (int v = 0; v < 8; ++v) {
            int co = m0 + waveM * 16 + v + 8 * hi;     // C/D layout: M = vgpr + 8*hi
            float val = acc[j][v] + bias[co];
            val = val > 0.f ? val : 0.f;
            out[((size_t)(bb * Cout + co) * H + yj) * W + xj] = (_Float16)val;
        }
    }
}

// ---------------------------------------------------------------- 2x2 maxpool (f16)
__global__ __launch_bounds__(256) void maxpool2(const _Float16* __restrict__ in,
                                                _Float16* __restrict__ out,
                                                int C, int Hin, int Win, int Bn)
{
    int Ho = Hin >> 1, Wo = Win >> 1;
    int total = Bn * C * Ho * Wo;
    int i = blockIdx.x * 256 + threadIdx.x;
    if (i >= total) return;
    int x = i % Wo; int tmp = i / Wo;
    int y = tmp % Ho; tmp /= Ho;          // tmp = b*C + c
    size_t base = ((size_t)tmp * Hin + 2 * y) * Win + 2 * x;
    float a = (float)in[base],        b = (float)in[base + 1];
    float c = (float)in[base + Win],  d = (float)in[base + Win + 1];
    out[i] = (_Float16)fmaxf(fmaxf(a, b), fmaxf(c, d));
}

// ---------------------------------------------------------------- per-position FC -> theta
__global__ __launch_bounds__(64) void fc_theta(const _Float16* __restrict__ g,
                                               const float* __restrict__ w1, const float* __restrict__ b1,
                                               const float* __restrict__ w2, const float* __restrict__ b2,
                                               float* __restrict__ theta)
{
    int row = blockIdx.x;        // b*256 + p
    int b   = row >> 8;
    int p   = row & 255;
    int t   = threadIdx.x;       // hidden unit
    __shared__ float hbuf[64];
    float acc = b1[t];
    const _Float16* gp = g + (size_t)b * 512 * 256 + p;
    const float* wr = w1 + t * 512;
    for (int c = 0; c < 512; ++c)
        acc += (float)gp[(size_t)c * 256] * wr[c];
    hbuf[t] = acc > 0.f ? acc : 0.f;
    __syncthreads();
    if (t < 6) {
        float o = b2[t];
        const float* w2r = w2 + t * 64;
        #pragma unroll 8
        for (int j = 0; j < 64; ++j) o += hbuf[j] * w2r[j];
        if (t == 1 || t == 3) o = fminf(fmaxf(o, -1e-4f), 1e-4f);
        theta[row * 6 + t] = o;
    }
}

// ---------------------------------------------------------------- grid-sample + stride-3 STN conv (fused)
__device__ __forceinline__ float padtap(const _Float16* __restrict__ fp, int yy, int xx) {
    if (yy >= 1 && yy <= 16 && xx >= 1 && xx <= 16)
        return (float)fp[(yy - 1) * 16 + (xx - 1)];
    return 0.f;
}

__global__ __launch_bounds__(256) void sample_stn(const _Float16* __restrict__ feat,
                                                  const float* __restrict__ theta,
                                                  const float* __restrict__ ws,
                                                  const float* __restrict__ bs,
                                                  float* __restrict__ logits)
{
    int row = blockIdx.x;        // b*256 + p
    int b = row >> 8, p = row & 255;
    int pi = p >> 4, pj = p & 15;
    const float* th = theta + row * 6;
    float t0 = th[0], t1 = th[1], t2 = th[2], t3 = th[3], t4 = th[4], t5 = th[5];

    float acc[NCLS];
    #pragma unroll
    for (int o = 0; o < NCLS; ++o) acc[o] = 0.f;

    for (int cc = 0; cc < 2; ++cc) {
        int c = threadIdx.x + cc * 256;
        const _Float16* fp = feat + ((size_t)b * 512 + c) * 256;
        float s[9];
        #pragma unroll
        for (int k = 0; k < 9; ++k) {
            float gx = -1.f + (2.f / 17.f) * (float)(pj + (k % 3));
            float gy = -1.f + (2.f / 17.f) * (float)(pi + (k / 3));
            float cx = t0 * gx + t1 * gy + t2;
            float cy = t3 * gx + t4 * gy + t5;
            float ix = (cx + 1.f) * 0.5f * 17.f;
            float iy = (cy + 1.f) * 0.5f * 17.f;
            float fx = floorf(ix), fy = floorf(iy);
            int ix0 = (int)fx, iy0 = (int)fy;
            float wx = ix - fx, wy = iy - fy;
            s[k] = padtap(fp, iy0,     ix0)     * (1.f - wx) * (1.f - wy)
                 + padtap(fp, iy0,     ix0 + 1) * wx         * (1.f - wy)
                 + padtap(fp, iy0 + 1, ix0)     * (1.f - wx) * wy
                 + padtap(fp, iy0 + 1, ix0 + 1) * wx         * wy;
        }
        #pragma unroll
        for (int o = 0; o < NCLS; ++o) {
            const float* wp = ws + ((size_t)o * 512 + c) * 9;
            float d = 0.f;
            #pragma unroll
            for (int k = 0; k < 9; ++k) d += s[k] * wp[k];
            acc[o] += d;
        }
    }

    __shared__ float red[256];
    for (int o = 0; o < NCLS; ++o) {
        red[threadIdx.x] = acc[o];
        __syncthreads();
        for (int s2 = 128; s2 > 0; s2 >>= 1) {
            if (threadIdx.x < s2) red[threadIdx.x] += red[threadIdx.x + s2];
            __syncthreads();
        }
        if (threadIdx.x == 0)
            logits[((size_t)b * NCLS + o) * 256 + p] = red[0] + bs[o];
        __syncthreads();
    }
}

// ---------------------------------------------------------------- softmax over (21*16*16) + spatial sums
__global__ __launch_bounds__(256) void softmax_out(const float* __restrict__ logits,
                                                   float* __restrict__ out)
{
    const int n = NCLS * 256;  // 5376
    int b = blockIdx.x;
    int t = threadIdx.x;
    const float* lg = logits + (size_t)b * n;
    __shared__ float red[256];

    float mx = -1e30f;
    for (int i = t; i < n; i += 256) mx = fmaxf(mx, lg[i]);
    red[t] = mx; __syncthreads();
    for (int s = 128; s > 0; s >>= 1) { if (t < s) red[t] = fmaxf(red[t], red[t + s]); __syncthreads(); }
    mx = red[0]; __syncthreads();

    float sum = 0.f;
    for (int i = t; i < n; i += 256) sum += expf(lg[i] - mx);
    red[t] = sum; __syncthreads();
    for (int s = 128; s > 0; s >>= 1) { if (t < s) red[t] += red[t + s]; __syncthreads(); }
    float inv = 1.f / red[0]; __syncthreads();

    float* probs = out + 2 * NCLS + (size_t)b * n;
    for (int i = t; i < n; i += 256) probs[i] = expf(lg[i] - mx) * inv;

    for (int o = 0; o < NCLS; ++o) {
        red[t] = expf(lg[o * 256 + t] - mx) * inv;
        __syncthreads();
        for (int s = 128; s > 0; s >>= 1) { if (t < s) red[t] += red[t + s]; __syncthreads(); }
        if (t == 0) out[b * NCLS + o] = red[0];
        __syncthreads();
    }
}

// ---------------------------------------------------------------- host orchestration
struct LayerCfg { int Cin, Cout, Hin; int pool; };
static const LayerCfg VGGL[13] = {
    {  3,  64, 512, 0}, { 64,  64, 512, 1},
    { 64, 128, 256, 0}, {128, 128, 256, 1},
    {128, 256, 128, 0}, {256, 256, 128, 0}, {256, 256, 128, 1},
    {256, 512,  64, 0}, {512, 512,  64, 0}, {512, 512,  64, 1},
    {512, 512,  32, 0}, {512, 512,  32, 0}, {512, 512,  32, 1}
};

extern "C" void kernel_launch(void* const* d_in, const int* in_sizes, int n_in,
                              void* d_out, int out_size, void* d_ws, size_t ws_size,
                              hipStream_t stream)
{
    (void)n_in; (void)out_size; (void)ws_size;
    const int Bn = 2;

    // ---- input index maps: dict-insertion order (x first) vs jax-sorted-key order
    int xi, vw[13], vb[13], lw[2], lb[2], f1w, f1b, f2w, f2b, stw, stb;
    if (in_sizes[0] == 2 * 3 * 512 * 512) {
        xi = 0;
        for (int i = 0; i < 13; ++i) { vw[i] = 1 + 2 * i; vb[i] = 2 + 2 * i; }
        for (int j = 0; j < 2;  ++j) { lw[j] = 27 + 2 * j; lb[j] = 28 + 2 * j; }
        f1w = 31; f1b = 32; f2w = 33; f2b = 34; stw = 35; stb = 36;
    } else {   // keys sorted: fc, loc, stn, vgg, then x
        f1w = 0; f1b = 1; f2w = 2; f2b = 3;
        for (int j = 0; j < 2;  ++j) { lw[j] = 4 + 2 * j; lb[j] = 5 + 2 * j; }
        stw = 8; stb = 9;
        for (int i = 0; i < 13; ++i) { vw[i] = 10 + 2 * i; vb[i] = 11 + 2 * i; }
        xi = 36;
    }

    // ---- carve workspace
    char* base = (char*)d_ws;
    size_t off = 0;
    auto carve = [&](size_t bytes) -> void* {
        void* p = base + off;
        off += (bytes + 255) & ~(size_t)255;
        return p;
    };

    _Float16* wh[15];                 // 13 vgg + 2 loc f16 weights, K reordered+padded
    int wCin[15], wCinp[15], wCout[15], widx[15];
    for (int i = 0; i < 15; ++i) {
        int Cin  = (i < 13) ? VGGL[i].Cin  : 512;
        int Cout = (i < 13) ? VGGL[i].Cout : 512;
        wCin[i]  = Cin;
        wCinp[i] = (Cin + 31) & ~31;
        wCout[i] = Cout;
        widx[i]  = (i < 13) ? vw[i] : lw[i - 13];
        wh[i] = (_Float16*)carve((size_t)Cout * 9 * wCinp[i] * sizeof(_Float16));
    }
    const size_t BIGBUF = (size_t)2 * 64 * 512 * 512;  // largest activation (halves)
    _Float16* bufA = (_Float16*)carve(BIGBUF * sizeof(_Float16));
    _Float16* bufB = (_Float16*)carve(BIGBUF * sizeof(_Float16));
    _Float16* feat = (_Float16*)carve((size_t)Bn * 512 * 256 * sizeof(_Float16));
    float* theta   = (float*)carve((size_t)Bn * 256 * 6 * sizeof(float));
    float* logits  = (float*)carve((size_t)Bn * NCLS * 256 * sizeof(float));

    // ---- convert weights (reorder+pad) + input to f16
    for (int i = 0; i < 15; ++i) {
        int total = wCout[i] * 9 * wCinp[i];
        cvt_wgt_reorder<<<(total + 255) / 256, 256, 0, stream>>>(
            (const float*)d_in[widx[i]], wh[i], wCin[i], wCinp[i], wCout[i]);
    }
    {
        int n = Bn * 3 * 512 * 512;
        cvt_f32_to_f16<<<(n + 255) / 256, 256, 0, stream>>>((const float*)d_in[xi], bufA, n);
    }

    auto launch_conv = [&](const _Float16* inp, const _Float16* w, const float* bi,
                           _Float16* o, int Cin, int Cout, int H) {
        int N = Bn * H * H;
        if (Cout % 128 == 0) {
            dim3 grid(N / 64, Cout / 128);
            conv3x3_wmma<128><<<grid, 256, 0, stream>>>(inp, w, bi, o, Cin, Cout, H, H);
        } else {
            dim3 grid(N / 64, Cout / 64);
            conv3x3_wmma<64><<<grid, 256, 0, stream>>>(inp, w, bi, o, Cin, Cout, H, H);
        }
    };

    // ---- VGG chain (ping-pong bufA/bufB)
    _Float16* cur = bufA;
    _Float16* nxt = bufB;
    for (int i = 0; i < 13; ++i) {
        int Cin = VGGL[i].Cin, Cout = VGGL[i].Cout, H = VGGL[i].Hin;
        launch_conv(cur, wh[i], (const float*)d_in[vb[i]], nxt, Cin, Cout, H);
        _Float16* tswap = cur; cur = nxt; nxt = tswap;
        if (VGGL[i].pool) {
            _Float16* dst = (i == 12) ? feat : nxt;
            int total = Bn * Cout * (H / 2) * (H / 2);
            maxpool2<<<(total + 255) / 256, 256, 0, stream>>>(cur, dst, Cout, H, H, Bn);
            if (i == 12) { cur = feat; }                    // nxt stays a free big buffer
            else         { _Float16* u = cur; cur = dst; nxt = u; }
        }
    }
    // cur == feat (B,512,16,16)

    // ---- localization convs
    _Float16* g0 = nxt;
    launch_conv(feat, wh[13], (const float*)d_in[lb[0]], g0, 512, 512, 16);
    _Float16* g1 = (g0 == bufA) ? bufB : bufA;
    launch_conv(g0, wh[14], (const float*)d_in[lb[1]], g1, 512, 512, 16);

    // ---- FC -> theta (with clipping), fused sample + STN conv, softmax + sums
    fc_theta<<<Bn * 256, 64, 0, stream>>>(g1,
                                          (const float*)d_in[f1w], (const float*)d_in[f1b],
                                          (const float*)d_in[f2w], (const float*)d_in[f2b],
                                          theta);
    sample_stn<<<Bn * 256, 256, 0, stream>>>(feat, theta,
                                             (const float*)d_in[stw], (const float*)d_in[stb],
                                             logits);
    softmax_out<<<Bn, 256, 0, stream>>>(logits, (float*)d_out);
}
